// MultiHeadAttention_29446295781713
// MI455X (gfx1250) — matrix operand order repaired and checked by reading, not compiled, not run on hardware
//
#include <hip/hip_runtime.h>
#include <hip/hip_bf16.h>

typedef _Float16 half_t;
typedef __attribute__((ext_vector_type(16))) _Float16 v16h;
typedef __attribute__((ext_vector_type(8)))  _Float16 v8h;
typedef __attribute__((ext_vector_type(8)))  float    v8f;
typedef __attribute__((ext_vector_type(4)))  unsigned int v4u;
typedef __attribute__((ext_vector_type(8)))  int      v8i;
typedef __attribute__((ext_vector_type(4)))  int      v4i;

union V16 { v16h v; v8h h[2]; };

static constexpr int DIMC  = 1024;   // model dim
static constexpr int MROWS = 8192;   // BS*QLEN
static constexpr int QLENC = 2048;
static constexpr int NHC   = 16;
static constexpr int DPHC  = 64;

__device__ __forceinline__ v8h ld8h(const half_t* p) {
    return *reinterpret_cast<const v8h*>(p);
}
__device__ __forceinline__ v8f zero_v8f() {
    v8f z = {0.f, 0.f, 0.f, 0.f, 0.f, 0.f, 0.f, 0.f};
    return z;
}
__device__ __forceinline__ v8f wmma16(v16h a, v16h b, v8f c) {
    // D = A(16x32 f16) * B(32x16 f16) + C(16x16 f32)
    return __builtin_amdgcn_wmma_f32_16x16x32_f16(
        /*neg_a=*/false, a, /*neg_b=*/false, b,
        /*c_mod=*/(short)0, c, /*reuse_a=*/false, /*reuse_b=*/false);
}

// ---------------------------------------------------------------------------
// TDM: 2-D tile load global->LDS. Descriptor per CDNA5 ISA
// (D# group0 + group1; groups 2/3 zero for 2-D). data_size = 2 bytes.
// Padding: after every 16 DWORDs (32 halves) insert 4 DWORDs (8 halves)
// -> reproduces LDS row stride of 40 halves for bank-conflict avoidance.
// This toolchain exposes the 6-arg builtin (g0,g1,g2,g3,g4,cpol).
// ---------------------------------------------------------------------------
__device__ __forceinline__ void tdm_load_2d(unsigned int lds_addr,
                                            const half_t* gaddr,
                                            unsigned int dim0,   // tensor width (elems)
                                            unsigned int dim1,   // tensor height (rows)
                                            unsigned int stride0,// row stride (elems)
                                            unsigned int tile0,  // tile width (elems)
                                            unsigned int tile1)  // tile height (rows)
{
    unsigned long long ga = (unsigned long long)(uintptr_t)gaddr;
    v4u g0;
    g0[0] = 1u;                                      // count=1, user descriptor
    g0[1] = lds_addr;                                // LDS byte address
    g0[2] = (unsigned int)(ga & 0xffffffffull);      // global_addr[31:0]
    g0[3] = (unsigned int)((ga >> 32) & 0x01fffffful) | (2u << 30); // addr[56:32] | type=2

    v8i g1;
    g1[0] = (int)((1u << 16) |            // data_size = 1 -> 2 bytes
                  (1u << 20) |            // pad_enable
                  (3u << 22) |            // pad_interval: 16 DWORDs
                  (3u << 25));            // pad_amount:   4 DWORDs
    g1[1] = (int)((dim0 & 0xffffu) << 16);                       // tensor_dim0 lo16
    g1[2] = (int)(((dim0 >> 16) & 0xffffu) | ((dim1 & 0xffffu) << 16));
    g1[3] = (int)(((dim1 >> 16) & 0xffffu) | ((tile0 & 0xffffu) << 16));
    g1[4] = (int)(tile1 & 0xffffu);                              // tile_dim1 (tile_dim2=0)
    g1[5] = (int)stride0;                                        // tensor_dim0_stride lo32
    g1[6] = 0;
    g1[7] = 0;

    v4i z4 = {0, 0, 0, 0};
    v8i z8 = {0, 0, 0, 0, 0, 0, 0, 0};
    __builtin_amdgcn_tensor_load_to_lds(g0, g1, z4, z4, z8, 0);
}

__device__ __forceinline__ unsigned int lds_off(const void* p) {
    return (unsigned int)(uintptr_t)p;   // low 32 bits of flat addr = LDS offset
}

// ---------------------------------------------------------------------------
// f32 -> f16 conversion (grid-stride)
// ---------------------------------------------------------------------------
__global__ void k_f32_to_f16(const float* __restrict__ in, half_t* __restrict__ out, int n) {
    int i = blockIdx.x * blockDim.x + threadIdx.x;
    int stride = gridDim.x * blockDim.x;
    for (; i < n; i += stride) out[i] = (half_t)in[i];
}

// f32 -> f16 with transpose: out[n*1024+k] = in[k*1024+n]  (weights, done once)
__global__ void k_f32_to_f16_T(const float* __restrict__ in, half_t* __restrict__ out) {
    int i = blockIdx.x * blockDim.x + threadIdx.x;   // output index
    if (i >= DIMC * DIMC) return;
    int n = i >> 10;
    int k = i & 1023;
    out[i] = (half_t)in[(size_t)k * DIMC + n];
}

// ---------------------------------------------------------------------------
// Tiled WMMA GEMM:  C[8192 x 1024] = A[8192 x 1024] * Wt^T + bias
// (Wt is the pre-transposed weight, row-major [n][k].)
// Block tile 128x64, K-step 32, 256 threads = 8 waves.
// Data movement: TDM double-buffered tile loads (wave 0 issues, TENSORcnt
// tracked), all waves compute WMMA from LDS.
// ---------------------------------------------------------------------------
template <bool F32OUT>
__device__ __forceinline__ void gemm_tile(const half_t* __restrict__ A,
                                          const half_t* __restrict__ Wt,
                                          const float*  __restrict__ bias,
                                          half_t* __restrict__ outH,
                                          float*  __restrict__ outF)
{
    __shared__ half_t As[2][128][40];   // 32 K-halves + 8 pad (filled by TDM pad)
    __shared__ half_t Bs[2][64][40];    // Bs[n][k]

    const int tid  = threadIdx.x;
    const int wave = tid >> 5;
    const int lane = tid & 31;
    const int g    = lane >> 4;
    const int ln   = lane & 15;
    const int m0   = blockIdx.x * 128;
    const int n0   = blockIdx.y * 64;
    const bool leader = (tid < 32);     // wave 0 issues the TDM descriptors

    v8f acc[4];
#pragma unroll
    for (int j = 0; j < 4; ++j) acc[j] = zero_v8f();

    if (leader) {
        tdm_load_2d(lds_off(&As[0][0][0]), A  + (size_t)m0 * DIMC, DIMC, MROWS, DIMC, 32, 128);
        tdm_load_2d(lds_off(&Bs[0][0][0]), Wt + (size_t)n0 * DIMC, DIMC, DIMC,  DIMC, 32, 64);
        __builtin_amdgcn_s_wait_tensorcnt(0);
    }
    __syncthreads();

    for (int k0 = 0; k0 < DIMC; k0 += 32) {
        const int cur = (k0 >> 5) & 1;
        const int nxt = cur ^ 1;

        // prefetch next K-step tiles into the other buffer (async, overlapped)
        if (leader && (k0 + 32 < DIMC)) {
            tdm_load_2d(lds_off(&As[nxt][0][0]), A  + (size_t)m0 * DIMC + k0 + 32,
                        DIMC, MROWS, DIMC, 32, 128);
            tdm_load_2d(lds_off(&Bs[nxt][0][0]), Wt + (size_t)n0 * DIMC + k0 + 32,
                        DIMC, DIMC, DIMC, 32, 64);
        }

        // --- compute from current buffer ---
        {
            const int mr = wave * 16 + ln;           // A fragment row
            V16 a;
            a.h[0] = *reinterpret_cast<const v8h*>(&As[cur][mr][g * 8]);
            a.h[1] = *reinterpret_cast<const v8h*>(&As[cur][mr][16 + g * 8]);
#pragma unroll
            for (int j = 0; j < 4; ++j) {
                V16 bf;
                bf.h[0] = *reinterpret_cast<const v8h*>(&Bs[cur][j * 16 + ln][g * 8]);
                bf.h[1] = *reinterpret_cast<const v8h*>(&Bs[cur][j * 16 + ln][16 + g * 8]);
                acc[j] = wmma16(a.v, bf.v, acc[j]);
            }
        }

        if (leader) __builtin_amdgcn_s_wait_tensorcnt(0);
        __syncthreads();
    }

    // --- epilogue: C element (M,N) -> lane (M/8)*16+N, VGPR M%8 ---
#pragma unroll
    for (int j = 0; j < 4; ++j) {
        int col = n0 + j * 16 + ln;
        float bv = bias[col];
#pragma unroll
        for (int v = 0; v < 8; ++v) {
            int row = m0 + wave * 16 + g * 8 + v;
            float val = acc[j][v] + bv;
            if (F32OUT) outF[(size_t)row * DIMC + col] = val;
            else        outH[(size_t)row * DIMC + col] = (half_t)val;
        }
    }
}

__global__ __launch_bounds__(256) void k_gemm_h(const half_t* __restrict__ A,
                                                const half_t* __restrict__ Wt,
                                                const float* __restrict__ bias,
                                                half_t* __restrict__ out) {
    gemm_tile<false>(A, Wt, bias, out, nullptr);
}
__global__ __launch_bounds__(256) void k_gemm_f(const half_t* __restrict__ A,
                                                const half_t* __restrict__ Wt,
                                                const float* __restrict__ bias,
                                                float* __restrict__ out) {
    gemm_tile<true>(A, Wt, bias, nullptr, out);
}

// ---------------------------------------------------------------------------
// Per-(row, head) L2 normalization: q = q/||q|| * scale ; k = k/||k||
// ---------------------------------------------------------------------------
__global__ void k_qk_norm(half_t* __restrict__ Q, half_t* __restrict__ K,
                          const float* __restrict__ scale_p) {
    int idx = blockIdx.x * blockDim.x + threadIdx.x;
    if (idx >= MROWS * NHC) return;
    int row = idx >> 4;
    int h   = idx & 15;
    half_t* q = Q + (size_t)row * DIMC + h * DPHC;
    half_t* k = K + (size_t)row * DIMC + h * DPHC;
    float s = scale_p[0];

    float ss = 0.f;
    for (int i = 0; i < DPHC; ++i) { float x = (float)q[i]; ss += x * x; }
    float inv = s / fmaxf(sqrtf(ss), 1e-12f);
    for (int i = 0; i < DPHC; ++i) q[i] = (half_t)((float)q[i] * inv);

    ss = 0.f;
    for (int i = 0; i < DPHC; ++i) { float x = (float)k[i]; ss += x * x; }
    inv = 1.0f / fmaxf(sqrtf(ss), 1e-12f);
    for (int i = 0; i < DPHC; ++i) k[i] = (half_t)((float)k[i] * inv);
}

// ---------------------------------------------------------------------------
// V transpose: Vt[bh][d][key] = V[b*2048+key][h*64+d]
// ---------------------------------------------------------------------------
__global__ void k_transpose_v(const half_t* __restrict__ V, half_t* __restrict__ Vt) {
    size_t i = (size_t)blockIdx.x * blockDim.x + threadIdx.x;
    size_t total = (size_t)64 * DPHC * QLENC;
    if (i >= total) return;
    int key = (int)(i & (QLENC - 1));
    int d   = (int)((i >> 11) & (DPHC - 1));
    int bh  = (int)(i >> 17);
    int b = bh >> 4, h = bh & 15;
    Vt[i] = V[((size_t)(b * QLENC + key)) * DIMC + h * DPHC + d];
}

// ---------------------------------------------------------------------------
// Flash attention: grid(32, 64) x 128 threads (4 waves). Wave owns 16 q rows.
// Per 32-key step: 4 WMMAs for S (K-dim 64), online softmax, P via LDS into
// A-fragment layout, 4 WMMAs for P@V into persistent f32 accumulators.
// ---------------------------------------------------------------------------
__global__ __launch_bounds__(128) void k_attn(const half_t* __restrict__ Q,
                                              const half_t* __restrict__ Km,
                                              const half_t* __restrict__ Vt,
                                              const int* __restrict__ mask,
                                              half_t* __restrict__ ctx)
{
    __shared__ half_t Pl[4][16][32];

    const int wave = threadIdx.x >> 5;
    const int lane = threadIdx.x & 31;
    const int g    = lane >> 4;
    const int ln   = lane & 15;
    const int bh   = blockIdx.y;
    const int b = bh >> 4, h = bh & 15;
    const int q0 = blockIdx.x * 64 + wave * 16;

    const half_t* qbase = Q  + ((size_t)(b * QLENC + q0)) * DIMC + h * DPHC;
    const half_t* kbase = Km + ((size_t)b * QLENC) * DIMC + h * DPHC;
    const half_t* vbase = Vt + (size_t)bh * DPHC * QLENC;
    const int*    mbase = mask + b * QLENC;

    // Q fragments (A-matrix, K-dim 64 -> two 16x32 fragments), loaded once.
    V16 a0, a1;
    {
        const half_t* qr = qbase + (size_t)ln * DIMC;
        a0.h[0] = ld8h(qr + g * 8);
        a0.h[1] = ld8h(qr + 16 + g * 8);
        a1.h[0] = ld8h(qr + 32 + g * 8);
        a1.h[1] = ld8h(qr + 48 + g * 8);
    }

    v8f acc[4];
#pragma unroll
    for (int j = 0; j < 4; ++j) acc[j] = zero_v8f();
    float mrow[8], lrow[8];
#pragma unroll
    for (int v = 0; v < 8; ++v) { mrow[v] = -1e30f; lrow[v] = 0.f; }

    for (int k0 = 0; k0 < QLENC; k0 += 32) {
        // prefetch K rows two steps ahead (-> global_prefetch_b8)
        if (k0 + 64 < QLENC) {
            __builtin_prefetch((const void*)(kbase + (size_t)(k0 + 64 + ln) * DIMC), 0, 1);
            __builtin_prefetch((const void*)(kbase + (size_t)(k0 + 80 + ln) * DIMC), 0, 1);
        }

        // ---- S tiles: keys [k0,k0+16) and [k0+16,k0+32) ----
        v8f s0 = zero_v8f(), s1 = zero_v8f();
        {
            const half_t* kr0 = kbase + (size_t)(k0 + ln) * DIMC;
            V16 bf0, bf1;
            bf0.h[0] = ld8h(kr0 + g * 8);
            bf0.h[1] = ld8h(kr0 + 16 + g * 8);
            bf1.h[0] = ld8h(kr0 + 32 + g * 8);
            bf1.h[1] = ld8h(kr0 + 48 + g * 8);
            s0 = wmma16(a0.v, bf0.v, s0);
            s0 = wmma16(a1.v, bf1.v, s0);
        }
        {
            const half_t* kr1 = kbase + (size_t)(k0 + 16 + ln) * DIMC;
            V16 bf0, bf1;
            bf0.h[0] = ld8h(kr1 + g * 8);
            bf0.h[1] = ld8h(kr1 + 16 + g * 8);
            bf1.h[0] = ld8h(kr1 + 32 + g * 8);
            bf1.h[1] = ld8h(kr1 + 48 + g * 8);
            s1 = wmma16(a0.v, bf0.v, s1);
            s1 = wmma16(a1.v, bf1.v, s1);
        }

        const int mk0 = mbase[k0 + ln];
        const int mk1 = mbase[k0 + 16 + ln];

        // ---- online softmax: row M=(g*8+v) lives in VGPR v across 16 lanes ----
        float p0[8], p1[8];
#pragma unroll
        for (int v = 0; v < 8; ++v) {
            float x0 = mk0 ? s0[v] : -1e30f;
            float x1 = mk1 ? s1[v] : -1e30f;
            float mx = fmaxf(x0, x1);
#pragma unroll
            for (int off = 1; off < 16; off <<= 1)
                mx = fmaxf(mx, __shfl_xor(mx, off, 32));
            float mnew  = fmaxf(mrow[v], mx);
            float alpha = __expf(mrow[v] - mnew);
            float e0 = mk0 ? __expf(x0 - mnew) : 0.f;
            float e1 = mk1 ? __expf(x1 - mnew) : 0.f;
            float rs = e0 + e1;
#pragma unroll
            for (int off = 1; off < 16; off <<= 1)
                rs += __shfl_xor(rs, off, 32);
            lrow[v] = lrow[v] * alpha + rs;
            mrow[v] = mnew;
            p0[v] = e0;
            p1[v] = e1;
#pragma unroll
            for (int j = 0; j < 4; ++j) acc[j][v] *= alpha;
        }

        // ---- route P (C layout) through LDS into A-fragment layout ----
#pragma unroll
        for (int v = 0; v < 8; ++v) {
            Pl[wave][g * 8 + v][ln]      = (half_t)p0[v];
            Pl[wave][g * 8 + v][16 + ln] = (half_t)p1[v];
        }
        asm volatile("s_wait_dscnt 0" ::: "memory");
        V16 ap;
        ap.h[0] = *reinterpret_cast<const v8h*>(&Pl[wave][ln][g * 8]);
        ap.h[1] = *reinterpret_cast<const v8h*>(&Pl[wave][ln][16 + g * 8]);

        // ---- ctx += P(16x32) @ V(32x64) : B from transposed V ----
#pragma unroll
        for (int j = 0; j < 4; ++j) {
            const half_t* vr = vbase + (size_t)(j * 16 + ln) * QLENC + k0 + g * 8;
            V16 bv;
            bv.h[0] = ld8h(vr);
            bv.h[1] = ld8h(vr + 16);
            acc[j] = wmma16(ap.v, bv.v, acc[j]);
        }
    }

    // ---- epilogue: normalize by row sums, store f16 ctx ----
    half_t* cb = ctx + ((size_t)(b * QLENC + q0)) * DIMC + h * DPHC;
#pragma unroll
    for (int v = 0; v < 8; ++v) {
        float inv = 1.0f / fmaxf(lrow[v], 1e-30f);
        int row = g * 8 + v;
#pragma unroll
        for (int j = 0; j < 4; ++j)
            cb[(size_t)row * DIMC + j * 16 + ln] = (half_t)(acc[j][v] * inv);
    }
}

// ---------------------------------------------------------------------------
extern "C" void kernel_launch(void* const* d_in, const int* in_sizes, int n_in,
                              void* d_out, int out_size, void* d_ws, size_t ws_size,
                              hipStream_t stream)
{
    (void)in_sizes; (void)n_in; (void)out_size; (void)ws_size;
    const float* x     = (const float*)d_in[0];
    const int*   mask  = (const int*)  d_in[1];
    const float* Wq    = (const float*)d_in[2];
    const float* bq    = (const float*)d_in[3];
    const float* Wk    = (const float*)d_in[4];
    const float* bk    = (const float*)d_in[5];
    const float* Wv    = (const float*)d_in[6];
    const float* bv    = (const float*)d_in[7];
    const float* scale = (const float*)d_in[8];
    const float* Wo    = (const float*)d_in[9];
    const float* bo    = (const float*)d_in[10];
    float* out = (float*)d_out;

    char* ws = (char*)d_ws;
    half_t* xh  = (half_t*)(ws + ((size_t)0  << 20));  // 16 MB
    half_t* wqh = (half_t*)(ws + ((size_t)16 << 20));  // 2 MB each (transposed)
    half_t* wkh = (half_t*)(ws + ((size_t)18 << 20));
    half_t* wvh = (half_t*)(ws + ((size_t)20 << 20));
    half_t* woh = (half_t*)(ws + ((size_t)22 << 20));
    half_t* Qh  = (half_t*)(ws + ((size_t)24 << 20));  // 16 MB each
    half_t* Kh  = (half_t*)(ws + ((size_t)40 << 20));
    half_t* Vh  = (half_t*)(ws + ((size_t)56 << 20));
    half_t* Vt  = (half_t*)(ws + ((size_t)72 << 20));
    half_t* Ch  = (half_t*)(ws + ((size_t)88 << 20));

    // 1) f32 -> f16 conversions (weights transposed once: Wt[n][k])
    k_f32_to_f16<<<4096, 256, 0, stream>>>(x, xh, MROWS * DIMC);
    k_f32_to_f16_T<<<4096, 256, 0, stream>>>(Wq, wqh);
    k_f32_to_f16_T<<<4096, 256, 0, stream>>>(Wk, wkh);
    k_f32_to_f16_T<<<4096, 256, 0, stream>>>(Wv, wvh);
    k_f32_to_f16_T<<<4096, 256, 0, stream>>>(Wo, woh);

    // 2) QKV projections (TDM-fed WMMA GEMM)
    dim3 gg(MROWS / 128, DIMC / 64);
    k_gemm_h<<<gg, 256, 0, stream>>>(xh, wqh, bq, Qh);
    k_gemm_h<<<gg, 256, 0, stream>>>(xh, wkh, bk, Kh);
    k_gemm_h<<<gg, 256, 0, stream>>>(xh, wvh, bv, Vh);

    // 3) cosine-attention normalization of Q (x scale) and K
    k_qk_norm<<<(MROWS * NHC) / 256, 256, 0, stream>>>(Qh, Kh, scale);

    // 4) V transpose for contiguous P@V B-fragments
    k_transpose_v<<<(64 * DPHC * QLENC) / 256, 256, 0, stream>>>(Vh, Vt);

    // 5) flash attention (WMMA, online softmax)
    k_attn<<<dim3(QLENC / 64, 4 * NHC), 128, 0, stream>>>(Qh, Kh, Vt, mask, Ch);

    // 6) output projection -> f32 out
    k_gemm_f<<<gg, 256, 0, stream>>>(Ch, woh, bo, out);
}
